// RGCN_41034117546481
// MI455X (gfx1250) — matrix operand order repaired and checked by reading, not compiled, not run on hardware
//
#include <hip/hip_runtime.h>
#include <hip/hip_bf16.h>

// ---------------- problem constants (match reference) ----------------
#define N_NODES 50000
#define N_EDGES 1000000
#define DIM     128
#define NR      16            // relations
#define NRELS   17            // relations + self slot
#define HSTRIDE (NRELS*DIM)   // 2176 columns in h
#define NLAYERS 6
#define NB      64
#define NK      128
#define LN_EPS  1e-5f

// ---------------- WMMA types (CDNA5 gfx1250, wave32) ----------------
typedef __attribute__((ext_vector_type(16))) __bf16 v16bf;
typedef __attribute__((ext_vector_type(8)))  float  v8f;

union BF16x16 { v16bf v; uint4 q[2]; unsigned short u[16]; };

__device__ __forceinline__ unsigned short f2bf(float f) {
    union { float f; unsigned u; } c; c.f = f;
    return (unsigned short)(c.u >> 16);          // truncate toward bf16
}
__device__ __forceinline__ float bf2f(unsigned short b) {
    union { unsigned u; float f; } c; c.u = ((unsigned)b) << 16;
    return c.f;
}

// ---------------- utility kernels ----------------
__global__ void zero_f32_kernel(float* __restrict__ p, long long n) {
    long long i = (long long)blockIdx.x * blockDim.x + threadIdx.x;
    if (i < n) p[i] = 0.0f;
}
__global__ void fill_ones_kernel(float* __restrict__ p, long long n) {
    long long i = (long long)blockIdx.x * blockDim.x + threadIdx.x;
    if (i < n) p[i] = 1.0f;
}
__global__ void degree_kernel(const int* __restrict__ edge_index, float* __restrict__ deg) {
    int e = blockIdx.x * blockDim.x + threadIdx.x;
    if (e < N_EDGES) atomicAdd(&deg[edge_index[N_EDGES + e]], 1.0f);
}
__global__ void invdeg_kernel(float* __restrict__ deg) {
    int v = blockIdx.x * blockDim.x + threadIdx.x;
    if (v < N_NODES) deg[v] = 1.0f / fmaxf(deg[v], 1.0f);
}

// ---------------- operand pre-packing (once per call / per layer) ----------------
// Wpack[l][c][k] bf16, k-major per output column c (c = r*128+f, r==16 -> Wself).
__global__ void pack_w_kernel(const float* __restrict__ Wrel,   // [L,16,128,128]
                              const float* __restrict__ Wself,  // [L,128,128]
                              unsigned short* __restrict__ Wp)  // [L,2176,128]
{
    long long t = (long long)blockIdx.x * blockDim.x + threadIdx.x;
    const long long total = (long long)NLAYERS * HSTRIDE * DIM;
    if (t >= total) return;
    int k = (int)(t & (DIM - 1));
    int c = (int)((t >> 7) % HSTRIDE);
    int l = (int)(t / ((long long)HSTRIDE * DIM));
    int rr = c >> 7;
    int f  = c & (DIM - 1);
    float v = (rr < NR)
        ? Wrel[(((size_t)l * NR + rr) * DIM + k) * DIM + f]
        : Wself[((size_t)l * DIM + k) * DIM + f];
    Wp[t] = f2bf(v);
}

// xb[v][k] bf16 from x f32 (pairs -> one dword store)
__global__ void pack_x_kernel(const float* __restrict__ x, unsigned* __restrict__ xb2) {
    long long i = (long long)blockIdx.x * blockDim.x + threadIdx.x;   // pair index
    const long long npairs = (long long)N_NODES * DIM / 2;
    if (i >= npairs) return;
    float2 p = *(const float2*)(x + i * 2);
    xb2[i] = ((unsigned)f2bf(p.y) << 16) | (unsigned)f2bf(p.x);
}

// ---------------- WMMA GEMM: h(bf16)[node, c] = xb @ Wpack_l ----------------
// Block: 128 threads = 4 waves. Wave tile: 64 rows x 32 cols (4x2 WMMA subtiles).
// Grid: (ceil(N/64), 2176/128) = (782, 17). All fragment loads are b128.
__global__ __launch_bounds__(128)
void rgcn_gemm_wmma_kernel(const unsigned short* __restrict__ xb,  // [N,128] bf16
                           const unsigned short* __restrict__ Wp,  // [2176,128] bf16
                           unsigned short* __restrict__ h)         // [N,2176] bf16
{
    const int lane = threadIdx.x & 31;
    const int wave = threadIdx.x >> 5;
    const int half = lane >> 4;       // 0: lanes 0-15, 1: lanes 16-31
    const int l16  = lane & 15;
    const int rowBase = blockIdx.x * 64;
    const int colBase = blockIdx.y * 128 + wave * 32;

    __builtin_prefetch(Wp + (size_t)colBase * DIM, 0, 1);  // global_prefetch toward L2

    v8f acc[4][2] = {};

    #pragma unroll
    for (int k0 = 0; k0 < DIM; k0 += 32) {
        // A fragments 16x32: elems 0..7 <- K=k0+half*8+e (16B contig),
        //                    elems 8..15 <- K=k0+16+half*8+e (16B contig at +32B)
        BF16x16 a[4];
        #pragma unroll
        for (int mt = 0; mt < 4; ++mt) {
            int row = rowBase + mt * 16 + l16;
            row = (row < N_NODES) ? row : (N_NODES - 1);   // clamp; stores guarded
            const unsigned short* ap = xb + (size_t)row * DIM + k0 + half * 8;
            a[mt].q[0] = *(const uint4*)(ap);
            a[mt].q[1] = *(const uint4*)(ap + 16);
        }
        // B fragments 32x16: elems 0..7 <- K=k0+half*8+e, elems 8..15 <- K=k0+16+half*8+e
        BF16x16 b[2];
        #pragma unroll
        for (int nt = 0; nt < 2; ++nt) {
            int c = colBase + nt * 16 + l16;
            const unsigned short* bp = Wp + (size_t)c * DIM + k0 + 8 * half;
            b[nt].q[0] = *(const uint4*)(bp);
            b[nt].q[1] = *(const uint4*)(bp + 16);
        }
        // 8 WMMAs per k-step
        #pragma unroll
        for (int mt = 0; mt < 4; ++mt)
            #pragma unroll
            for (int nt = 0; nt < 2; ++nt)
                acc[mt][nt] = __builtin_amdgcn_wmma_f32_16x16x32_bf16(
                    false, a[mt].v, false, b[nt].v,
                    (short)0, acc[mt][nt], false, false);
    }

    // store D (C layout: VGPR i -> M = i + 8*half), f32 -> bf16
    #pragma unroll
    for (int mt = 0; mt < 4; ++mt) {
        #pragma unroll
        for (int nt = 0; nt < 2; ++nt) {
            int c = colBase + nt * 16 + l16;
            #pragma unroll
            for (int i = 0; i < 8; ++i) {
                int row = rowBase + mt * 16 + half * 8 + i;
                if (row < N_NODES)
                    h[(size_t)row * HSTRIDE + c] = f2bf(acc[mt][nt][i]);
            }
        }
    }
}

// ---------------- per-edge gather + scatter-add into agg (f32, L2-resident) ----------------
// 32 threads per edge; each thread reads 4 bf16 (uint2) and issues 4 f32 atomics.
__global__ void scatter_kernel(const unsigned short* __restrict__ h,
                               const int* __restrict__ edge_index,
                               const int* __restrict__ edge_type,
                               float* __restrict__ agg)
{
    long long gid = (long long)blockIdx.x * blockDim.x + threadIdx.x;
    int e = (int)(gid >> 5);
    int j = (int)(gid & 31);
    if (e >= N_EDGES) return;
    int src = edge_index[e];
    int dst = edge_index[N_EDGES + e];
    int r   = edge_type[e];
    const uint2* hp = (const uint2*)(h + (size_t)src * HSTRIDE + r * DIM + j * 4);
    uint2 pk = *hp;
    float* ap = agg + (size_t)dst * DIM + j * 4;
    atomicAdd(ap + 0, bf2f((unsigned short)(pk.x & 0xffffu)));
    atomicAdd(ap + 1, bf2f((unsigned short)(pk.x >> 16)));
    atomicAdd(ap + 2, bf2f((unsigned short)(pk.y & 0xffffu)));
    atomicAdd(ap + 3, bf2f((unsigned short)(pk.y >> 16)));
}

// ---------------- combine: out = relu(LN(agg*inv_deg + self)) + x_old ; one wave32/node ----
__global__ void combine_kernel(const float* __restrict__ agg,
                               const unsigned short* __restrict__ h,   // self slice @ col 2048
                               const float* __restrict__ inv_deg,
                               const float* __restrict__ g,
                               const float* __restrict__ bta,
                               const float* __restrict__ x_old,
                               float* __restrict__ x_new)
{
    int gthread = blockIdx.x * blockDim.x + threadIdx.x;
    int v    = gthread >> 5;
    int lane = gthread & 31;
    if (v >= N_NODES) return;

    float idv = inv_deg[v];
    int c0 = lane * 4;
    float4 a = *(const float4*)(agg + (size_t)v * DIM + c0);
    const uint2 hp = *(const uint2*)(h + (size_t)v * HSTRIDE + NR * DIM + c0);
    float val[4];
    val[0] = a.x * idv + bf2f((unsigned short)(hp.x & 0xffffu));
    val[1] = a.y * idv + bf2f((unsigned short)(hp.x >> 16));
    val[2] = a.z * idv + bf2f((unsigned short)(hp.y & 0xffffu));
    val[3] = a.w * idv + bf2f((unsigned short)(hp.y >> 16));

    float s = val[0] + val[1] + val[2] + val[3];
    #pragma unroll
    for (int m = 16; m >= 1; m >>= 1) s += __shfl_xor(s, m, 32);
    float mu = s * (1.0f / 128.0f);

    float q = 0.0f;
    #pragma unroll
    for (int i = 0; i < 4; ++i) { float d = val[i] - mu; q += d * d; }
    #pragma unroll
    for (int m = 16; m >= 1; m >>= 1) q += __shfl_xor(q, m, 32);
    float rs = rsqrtf(q * (1.0f / 128.0f) + LN_EPS);

    float4 xo = *(const float4*)(x_old + (size_t)v * DIM + c0);
    float4 go = *(const float4*)(g + c0);
    float4 bo = *(const float4*)(bta + c0);
    float4 r;
    r.x = fmaxf(go.x * (val[0] - mu) * rs + bo.x, 0.0f) + xo.x;
    r.y = fmaxf(go.y * (val[1] - mu) * rs + bo.y, 0.0f) + xo.y;
    r.z = fmaxf(go.z * (val[2] - mu) * rs + bo.z, 0.0f) + xo.z;
    r.w = fmaxf(go.w * (val[3] - mu) * rs + bo.w, 0.0f) + xo.w;
    *(float4*)(x_new + (size_t)v * DIM + c0) = r;
}

// ---------------- final scoring: out[b,k] = sum_d x[s]*rel_emb[r]*x[t]; one wave32/triple ----
__global__ void score_kernel(const float* __restrict__ x,
                             const int* __restrict__ batch,     // [B*K, 3]
                             const float* __restrict__ rel_emb, // [R, 128]
                             float* __restrict__ out)
{
    int gthread = blockIdx.x * blockDim.x + threadIdx.x;
    int idx  = gthread >> 5;
    int lane = gthread & 31;
    if (idx >= NB * NK) return;
    int s = batch[idx * 3 + 0];
    int t = batch[idx * 3 + 1];
    int r = batch[idx * 3 + 2];
    int c0 = lane * 4;
    float4 sv = *(const float4*)(x + (size_t)s * DIM + c0);
    float4 tv = *(const float4*)(x + (size_t)t * DIM + c0);
    float4 rv = *(const float4*)(rel_emb + (size_t)r * DIM + c0);
    float acc = sv.x * rv.x * tv.x + sv.y * rv.y * tv.y
              + sv.z * rv.z * tv.z + sv.w * rv.w * tv.w;
    #pragma unroll
    for (int m = 16; m >= 1; m >>= 1) acc += __shfl_xor(acc, m, 32);
    if (lane == 0) out[idx] = acc;
}

// ---------------- workspace layout ----------------
static constexpr size_t SZ_X   = (size_t)N_NODES * DIM * 4;          // 25.6 MB
static constexpr size_t OFF_DEG = 0;                                 // 50000 f32
static constexpr size_t OFF_X0  = 200192;
static constexpr size_t OFF_X1  = OFF_X0 + SZ_X;
static constexpr size_t OFF_AGG = OFF_X1 + SZ_X;
static constexpr size_t OFF_XB  = OFF_AGG + SZ_X;                    // 12.8 MB bf16
static constexpr size_t OFF_WP  = OFF_XB + (size_t)N_NODES * DIM * 2;// 3.34 MB bf16
static constexpr size_t OFF_H   = OFF_WP + (size_t)NLAYERS * HSTRIDE * DIM * 2; // 217.6 MB

extern "C" void kernel_launch(void* const* d_in, const int* in_sizes, int n_in,
                              void* d_out, int out_size, void* d_ws, size_t ws_size,
                              hipStream_t stream) {
    const int*   edge_index = (const int*)d_in[0];
    const int*   edge_type  = (const int*)d_in[1];
    const int*   batch      = (const int*)d_in[2];
    const float* Wrel       = (const float*)d_in[3];  // [L,R,D,D]
    const float* Wself      = (const float*)d_in[4];  // [L,D,D]
    const float* ln_g       = (const float*)d_in[5];  // [L,D]
    const float* ln_b       = (const float*)d_in[6];  // [L,D]
    const float* rel_emb    = (const float*)d_in[7];  // [R,D]
    float* out = (float*)d_out;

    char* ws = (char*)d_ws;
    float*          deg = (float*)(ws + OFF_DEG);
    float*          x0  = (float*)(ws + OFF_X0);
    float*          x1  = (float*)(ws + OFF_X1);
    float*          agg = (float*)(ws + OFF_AGG);
    unsigned short* xb  = (unsigned short*)(ws + OFF_XB);
    unsigned short* Wp  = (unsigned short*)(ws + OFF_WP);
    unsigned short* h   = (unsigned short*)(ws + OFF_H);

    const long long nfeat = (long long)N_NODES * DIM;

    // degree -> inv_deg (in-place)
    zero_f32_kernel<<<(N_NODES + 255) / 256, 256, 0, stream>>>(deg, N_NODES);
    degree_kernel<<<(N_EDGES + 255) / 256, 256, 0, stream>>>(edge_index, deg);
    invdeg_kernel<<<(N_NODES + 255) / 256, 256, 0, stream>>>(deg);

    // pack all layers' weights to bf16 (k-major per output column)
    {
        long long total = (long long)NLAYERS * HSTRIDE * DIM;
        pack_w_kernel<<<(int)((total + 255) / 256), 256, 0, stream>>>(Wrel, Wself, Wp);
    }

    // x = ones
    fill_ones_kernel<<<(int)((nfeat + 255) / 256), 256, 0, stream>>>(x0, nfeat);

    float* xc = x0;
    float* xn = x1;
    dim3 gemm_grid((N_NODES + 63) / 64, HSTRIDE / 128);  // (782, 17)

    for (int l = 0; l < NLAYERS; ++l) {
        pack_x_kernel<<<(int)((nfeat / 2 + 255) / 256), 256, 0, stream>>>(xc, (unsigned*)xb);
        rgcn_gemm_wmma_kernel<<<gemm_grid, 128, 0, stream>>>(
            xb, Wp + (size_t)l * HSTRIDE * DIM, h);
        zero_f32_kernel<<<(int)((nfeat + 255) / 256), 256, 0, stream>>>(agg, nfeat);
        {
            long long nthreads = (long long)N_EDGES * 32;
            scatter_kernel<<<(int)((nthreads + 255) / 256), 256, 0, stream>>>(
                h, edge_index, edge_type, agg);
        }
        combine_kernel<<<(N_NODES * 32 + 255) / 256, 256, 0, stream>>>(
            agg, h, deg, ln_g + (size_t)l * DIM, ln_b + (size_t)l * DIM, xc, xn);
        float* t = xc; xc = xn; xn = t;
    }

    // after 6 swaps the final features are back in x0 (== xc)
    score_kernel<<<(NB * NK * 32 + 255) / 256, 256, 0, stream>>>(xc, batch, rel_emb, out);
}
